// TimeAwareAffinityPredictor_75883482186257
// MI455X (gfx1250) — matrix operand order repaired and checked by reading, not compiled, not run on hardware
//
#include <hip/hip_runtime.h>
#include <hip/hip_bf16.h>

#define NLIG  16384
#define NPROT 131072
#define NTOT  (NLIG + NPROT)      // 147456
#define NE    2097152
#define NB    256
#define HD    64

typedef __attribute__((ext_vector_type(16))) _Float16     v16h;
typedef __attribute__((ext_vector_type(8)))  float        v8f;
typedef __attribute__((ext_vector_type(4)))  unsigned int u32x4;
typedef __attribute__((ext_vector_type(8)))  unsigned int u32x8;

// ---------------- small utility kernels ----------------

__global__ __launch_bounds__(256) void fill_zero4_k(float4* p, long long n4) {
    long long i = (long long)blockIdx.x * 256 + threadIdx.x;
    if (i < n4) p[i] = make_float4(0.f, 0.f, 0.f, 0.f);
}

// t_emb sin/cos table: emb[b, j] ; 256*64 threads
__global__ __launch_bounds__(256) void time_embed_k(const float* __restrict__ t,
                                                    float* __restrict__ emb) {
    int idx = blockIdx.x * 256 + threadIdx.x;       // 16384
    int b = idx >> 6, j = idx & 63;
    float tv = t[b];
    float out;
    const float c = -logf(10000.0f) / 31.0f;        // half-1 = 31
    if (j < 32) {
        out = sinf(tv * expf((float)j * c));
    } else {
        out = cosf(tv * expf((float)(j - 32) * c));
    }
    emb[idx] = out;
}

// generic [rows x 64] @ [64 x 64] + bias, optional SiLU (scalar; tiny matrices)
__global__ __launch_bounds__(256) void mlp64_k(const float* __restrict__ in,
                                               const float* __restrict__ W,
                                               const float* __restrict__ bias,
                                               float* __restrict__ out,
                                               int rows, int applySilu) {
    int idx = blockIdx.x * 256 + threadIdx.x;
    if (idx >= rows * 64) return;
    int r = idx >> 6, h = idx & 63;
    float acc = bias[h];
    const float* ri = in + (size_t)r * 64;
    #pragma unroll 8
    for (int k = 0; k < 64; ++k) acc += ri[k] * W[k * 64 + h];
    if (applySilu) acc = acc / (1.0f + expf(-acc));   // x*sigmoid(x)
    out[idx] = acc;
}

// node feature init: lig: feat@W_lig + b_lig + temb[batch] ; prot: feat@W_prot + b_prot
__global__ __launch_bounds__(256) void init_nodes_k(const float* __restrict__ ligf,
                                                    const float* __restrict__ protf,
                                                    const float* __restrict__ Wlig,
                                                    const float* __restrict__ blig,
                                                    const float* __restrict__ Wprot,
                                                    const float* __restrict__ bprot,
                                                    const float* __restrict__ temb,
                                                    const int* __restrict__ ligb,
                                                    float* __restrict__ x) {
    long long idx = (long long)blockIdx.x * 256 + threadIdx.x;
    if (idx >= (long long)NTOT * HD) return;
    int i = (int)(idx >> 6), h = (int)(idx & 63);
    float acc;
    if (i < NLIG) {
        acc = blig[h] + temb[(size_t)ligb[i] * HD + h];
        const float* f = ligf + (size_t)i * 15;
        #pragma unroll
        for (int k = 0; k < 15; ++k) acc += f[k] * Wlig[k * HD + h];
    } else {
        int p = i - NLIG;
        acc = bprot[h];
        const float* f = protf + (size_t)p * 8;
        #pragma unroll
        for (int k = 0; k < 8; ++k) acc += f[k] * Wprot[k * HD + h];
    }
    x[idx] = acc;
}

// pack the 6 relational weight matrices as f16, transposed: out[w][n*64+k] = W_w[k*64+n]
// layout: layer l -> [2l]=Wrel^T, [2l+1]=Wroot^T (contiguous 16KB per layer for TDM)
__global__ __launch_bounds__(256) void prep_w16_k(const float* w0, const float* w1,
                                                  const float* w2, const float* w3,
                                                  const float* w4, const float* w5,
                                                  _Float16* out) {
    int idx = blockIdx.x * 256 + threadIdx.x;       // 6*4096
    if (idx >= 6 * 4096) return;
    int w = idx >> 12, r = idx & 4095;
    int n = r >> 6, k = r & 63;
    const float* W = (w == 0) ? w0 : (w == 1) ? w1 : (w == 2) ? w2
                    : (w == 3) ? w3 : (w == 4) ? w4 : w5;
    out[idx] = (_Float16)W[k * 64 + n];
}

// ---------------- edge scatter-add (the bandwidth/atomic bottleneck) ----------------
// 16 threads per edge, one float4 each -> half-wave reads a full 256B source row coalesced.
// x and agg (37.7 MB each) stay L2-resident; edge stream prefetched ahead.
__global__ __launch_bounds__(256) void scatter_edges_k(const int* __restrict__ eidx,
                                                       const float* __restrict__ x,
                                                       float* agg) {
    long long idx = (long long)blockIdx.x * 256 + threadIdx.x;  // E*16
    if (idx >= (long long)NE * 16) return;
    int e = (int)(idx >> 4);
    int q = (int)(idx & 15);
    if (q == 0 && e + 8192 < NE) {       // pull index stream ahead (global_prefetch_b8)
        __builtin_prefetch(eidx + e + 8192, 0, 3);
        __builtin_prefetch(eidx + NE + e + 8192, 0, 3);
    }
    int src = eidx[e];
    int dst = eidx[NE + e];
    const float4 v = *(const float4*)(x + (size_t)src * HD + q * 4);
    float* d = agg + (size_t)dst * HD + q * 4;
    atomicAdd(d + 0, v.x);
    atomicAdd(d + 1, v.y);
    atomicAdd(d + 2, v.z);
    atomicAdd(d + 3, v.w);
}

// ---------------- graph-conv dense update via WMMA ----------------
// agg (read, then overwritten in place): output row m depends only on input rows m,
// which are fully register-resident before the stores -> in-place is safe.
// out = agg @ Wrel + brel + xin @ Wroot
// Weights (16 KB: Wrel^T || Wroot^T in f16) staged into LDS by the Tensor Data Mover.
__global__ __launch_bounds__(256) void conv_gemm_k(float* agg,                    // in/out
                                                   const float* __restrict__ xin,
                                                   const _Float16* __restrict__ wT, // 2*4096 halves
                                                   const float* __restrict__ brel) {
    __shared__ _Float16 sW[2 * 4096];   // [0..4095]=Wrel^T, [4096..8191]=Wroot^T
    __shared__ float    sB[64];
    int tid = threadIdx.x;

    if (tid == 0) {
        // Tensor DMA descriptor (ISA 8.3/8.4): 1 x 8192 tile of 2-byte elements.
        unsigned long long ga = (unsigned long long)(uintptr_t)wT;
        unsigned lds = (unsigned)(uintptr_t)(&sW[0]);
        u32x4 g0;
        g0[0] = 1u;                                             // count=1, user mode
        g0[1] = lds;                                            // lds_addr [63:32]
        g0[2] = (unsigned)(ga & 0xffffffffu);                   // global_addr [95:64]
        g0[3] = (unsigned)((ga >> 32) & 0x01ffffffu) | (2u << 30); // addr[56:32] | type=2
        u32x8 g1;
        g1[0] = 1u << 16;            // workgroup_mask=0, data_size=1 (2 bytes)
        g1[1] = 0x2000u << 16;       // tensor_dim0 = 8192 (low 16 bits at [63:48])
        g1[2] = 1u << 16;            // tensor_dim0 hi = 0 ; tensor_dim1 = 1
        g1[3] = 0x2000u << 16;       // tensor_dim1 hi = 0 ; tile_dim0 = 8192
        g1[4] = 0u;                  // tile_dim1 = 0 (unused), tile_dim2 = 0
        g1[5] = 8192u;               // tensor_dim0_stride = 8192 (low 32)
        g1[6] = 0u;                  // stride hi / dim1_stride lo
        g1[7] = 0u;
        asm volatile("tensor_load_to_lds %0, %1" :: "s"(g0), "s"(g1) : "memory");
        __builtin_amdgcn_s_wait_tensorcnt(0);
    }
    if (tid < 64) sB[tid] = brel[tid];
    __syncthreads();

    int wave = tid >> 5, lane = tid & 31;
    int rowBase = blockIdx.x * 128 + wave * 16;
    int row = rowBase + (lane & 15);
    int chalf = (lane < 16) ? 0 : 8;       // A-matrix K interleave per ISA table

    // Load + convert A tiles (f32 -> f16) for both operand streams, K=0..31 and 32..63
    v16h aAgg[2], aX[2];
    const float* ra = agg + (size_t)row * HD;
    const float* rx = xin + (size_t)row * HD;
    #pragma unroll
    for (int kt = 0; kt < 2; ++kt) {
        int c0 = chalf + kt * 32;
        #pragma unroll
        for (int j = 0; j < 8; ++j) {
            aAgg[kt][j]     = (_Float16)ra[c0 + j];
            aAgg[kt][j + 8] = (_Float16)ra[c0 + 16 + j];
            aX[kt][j]       = (_Float16)rx[c0 + j];
            aX[kt][j + 8]   = (_Float16)rx[c0 + 16 + j];
        }
    }

    int nl = lane & 15;
    int mBase = rowBase + ((lane < 16) ? 0 : 8);

    #pragma unroll
    for (int ct = 0; ct < 4; ++ct) {                 // 4 column tiles of 16
        int n = ct * 16 + nl;
        float bv = sB[n];
        v8f c;
        #pragma unroll
        for (int j = 0; j < 8; ++j) c[j] = bv;       // bias broadcast into C

        #pragma unroll
        for (int kt = 0; kt < 2; ++kt) {
            int koff = kt * 32 + ((lane < 16) ? 0 : 8);
            v16h bRel, bRoot;
            #pragma unroll
            for (int j = 0; j < 16; ++j) {
                int kk = koff + j + ((j < 8) ? 0 : 8);
                bRel[j]  = sW[n * 64 + kk];
                bRoot[j] = sW[4096 + n * 64 + kk];
            }
            c = __builtin_amdgcn_wmma_f32_16x16x32_f16(false, aAgg[kt], false, bRel,
                                                       (short)0, c, false, false);
            c = __builtin_amdgcn_wmma_f32_16x16x32_f16(false, aX[kt], false, bRoot,
                                                       (short)0, c, false, false);
        }
        // C layout: VGPR j -> M = mBase + j, N = n
        #pragma unroll
        for (int j = 0; j < 8; ++j)
            agg[(size_t)(mBase + j) * HD + n] = c[j];
    }
}

// ---------------- readout ----------------

__global__ __launch_bounds__(256) void pool_accum_k(const float* __restrict__ x,
                                                    const int* __restrict__ ligb,
                                                    float* pooled, float* cnt) {
    int idx = blockIdx.x * 256 + threadIdx.x;       // NLIG*64
    if (idx >= NLIG * 64) return;
    int i = idx >> 6, h = idx & 63;
    int b = ligb[i];
    atomicAdd(&pooled[b * 64 + h], x[(size_t)i * HD + h]);
    if (h == 0) atomicAdd(&cnt[b], 1.0f);
}

__global__ __launch_bounds__(256) void pool_mean_k(float* pooled, const float* cnt) {
    int idx = blockIdx.x * 256 + threadIdx.x;       // 256*64
    if (idx >= NB * 64) return;
    pooled[idx] /= fmaxf(cnt[idx >> 6], 1.0f);
}

__global__ __launch_bounds__(256) void head_out_k(const float* __restrict__ hid,
                                                  const float* __restrict__ Wro2,
                                                  const float* __restrict__ bro2,
                                                  float* __restrict__ out) {
    int b = threadIdx.x;                            // 256 threads, 1 block
    float acc = bro2[0];
    #pragma unroll 8
    for (int h = 0; h < 64; ++h) acc += hid[b * 64 + h] * Wro2[h];
    out[b] = acc;
}

// ---------------- host orchestration ----------------

extern "C" void kernel_launch(void* const* d_in, const int* in_sizes, int n_in,
                              void* d_out, int out_size, void* d_ws, size_t ws_size,
                              hipStream_t stream) {
    const float* lig_feat  = (const float*)d_in[1];
    const float* prot_feat = (const float*)d_in[3];
    const float* t         = (const float*)d_in[4];
    const int*   lig_batch = (const int*)d_in[5];
    const int*   edge_idx  = (const int*)d_in[7];
    const float* W_lig  = (const float*)d_in[8];
    const float* b_lig  = (const float*)d_in[9];
    const float* W_prot = (const float*)d_in[10];
    const float* b_prot = (const float*)d_in[11];
    const float* Wt1 = (const float*)d_in[12];
    const float* bt1 = (const float*)d_in[13];
    const float* Wt2 = (const float*)d_in[14];
    const float* bt2 = (const float*)d_in[15];
    const float* Wrel[3]  = {(const float*)d_in[16], (const float*)d_in[19], (const float*)d_in[22]};
    const float* brel[3]  = {(const float*)d_in[17], (const float*)d_in[20], (const float*)d_in[23]};
    const float* Wroot[3] = {(const float*)d_in[18], (const float*)d_in[21], (const float*)d_in[24]};
    const float* Wro1 = (const float*)d_in[25];
    const float* bro1 = (const float*)d_in[26];
    const float* Wro2 = (const float*)d_in[27];
    const float* bro2 = (const float*)d_in[28];
    float* out = (float*)d_out;

    // workspace carve-up (~76 MB)
    float* bufA   = (float*)d_ws;                 // N*64
    float* bufB   = bufA + (size_t)NTOT * HD;     // N*64
    float* emb    = bufB + (size_t)NTOT * HD;     // 256*64
    float* h1     = emb  + NB * HD;
    float* temb   = h1   + NB * HD;
    float* pooled = temb + NB * HD;               // 256*64
    float* cnt    = pooled + NB * HD;             // 256
    float* hid    = cnt + NB;                     // 256*64
    _Float16* w16 = (_Float16*)(hid + NB * HD);   // 6 * 4096 halves

    // 1) time embedding MLP (tiny)
    time_embed_k<<<64, 256, 0, stream>>>(t, emb);
    mlp64_k<<<64, 256, 0, stream>>>(emb, Wt1, bt1, h1, NB, 1);
    mlp64_k<<<64, 256, 0, stream>>>(h1, Wt2, bt2, temb, NB, 0);

    // 2) node init -> bufA
    long long ntot64 = (long long)NTOT * HD;
    init_nodes_k<<<(int)((ntot64 + 255) / 256), 256, 0, stream>>>(
        lig_feat, prot_feat, W_lig, b_lig, W_prot, b_prot, temb, lig_batch, bufA);

    // 3) pack relational weights as f16 transposed (layer-contiguous for TDM)
    prep_w16_k<<<96, 256, 0, stream>>>(Wrel[0], Wroot[0], Wrel[1], Wroot[1],
                                       Wrel[2], Wroot[2], w16);

    // 4) three graph-conv layers (ping-pong bufA <-> bufB)
    float* xcur = bufA;
    float* xagg = bufB;
    for (int l = 0; l < 3; ++l) {
        fill_zero4_k<<<(int)(ntot64 / 4 / 256), 256, 0, stream>>>((float4*)xagg, ntot64 / 4);
        scatter_edges_k<<<(int)(((long long)NE * 16) / 256), 256, 0, stream>>>(
            edge_idx, xcur, xagg);
        conv_gemm_k<<<NTOT / 128, 256, 0, stream>>>(
            xagg, xcur, w16 + (size_t)(2 * l) * 4096, brel[l]);
        float* tmp = xcur; xcur = xagg; xagg = tmp;   // xcur now holds new features
    }

    // 5) mean-pool ligand nodes per batch
    fill_zero4_k<<<((NB * HD + NB) / 4 + 255) / 256, 256, 0, stream>>>(
        (float4*)pooled, (NB * HD + NB) / 4);
    pool_accum_k<<<(NLIG * 64) / 256, 256, 0, stream>>>(xcur, lig_batch, pooled, cnt);
    pool_mean_k<<<(NB * HD) / 256, 256, 0, stream>>>(pooled, cnt);

    // 6) readout head
    mlp64_k<<<64, 256, 0, stream>>>(pooled, Wro1, bro1, hid, NB, 1);
    head_out_k<<<1, 256, 0, stream>>>(hid, Wro2, bro2, out);
}
